// TransformerBlock_67577015435485
// MI455X (gfx1250) — matrix operand order repaired and checked by reading, not compiled, not run on hardware
//
#include <hip/hip_runtime.h>
#include <hip/hip_bf16.h>
#include <stdint.h>

typedef __attribute__((ext_vector_type(16))) __bf16 v16bf;
typedef __attribute__((ext_vector_type(8)))  float  v8f;
typedef __attribute__((ext_vector_type(4)))  int    v4i;

#define NNODES 50000
#define NEDGES 400000
#define DEMB   512
#define NH     8

#if __has_builtin(__builtin_amdgcn_global_load_async_to_lds_b128) && \
    __has_builtin(__builtin_amdgcn_s_wait_asynccnt)
#define USE_ASYNC_LDS 1
#else
#define USE_ASYNC_LDS 0
#endif

#if USE_ASYNC_LDS
typedef __attribute__((address_space(1))) v4i as1_v4i;   // global int4
typedef __attribute__((address_space(3))) v4i as3_v4i;   // LDS int4
__device__ __forceinline__ void async_cp16(const void* g, void* l) {
  // global -> LDS, 16B per lane, tracked on ASYNCcnt, no VGPR data path.
  // LDS generic->as3: low 32 bits of the flat address are the LDS offset.
  __builtin_amdgcn_global_load_async_to_lds_b128(
      (as1_v4i*)(uintptr_t)g, (as3_v4i*)(uint32_t)(uintptr_t)l, 0, 0);
}
#endif

// ---------------------------------------------------------------- utilities
__global__ void zero_f32(float* __restrict__ p, int n) {
  int i = blockIdx.x * blockDim.x + threadIdx.x;
  if (i < n) p[i] = 0.f;
}

__global__ void cvt_f32_bf16(const float* __restrict__ src, __bf16* __restrict__ dst, int n) {
  int i = blockIdx.x * blockDim.x + threadIdx.x;
  if (i < n) dst[i] = (__bf16)src[i];
}

__global__ void relu_cvt_bf16(const float* __restrict__ src, __bf16* __restrict__ dst, int n) {
  int i = blockIdx.x * blockDim.x + threadIdx.x;
  if (i < n) {
    float v = src[i];
    dst[i] = (__bf16)(v > 0.f ? v : 0.f);
  }
}

// WT[n][k] = W[k][n], fp32 -> bf16   (W is [K,N] row-major)
__global__ void cvt_transpose_bf16(const float* __restrict__ W, __bf16* __restrict__ WT,
                                   int K, int N) {
  int i = blockIdx.x * blockDim.x + threadIdx.x;
  if (i < K * N) {
    int n = i / K, k = i % K;
    WT[i] = (__bf16)W[(size_t)k * N + n];
  }
}

// WqkvT[c][k] (c: 0..255=K, 256..511=Q, 512..767=V), biases concat fp32[768]
__global__ void cvt_qkv_weightsT(const float* __restrict__ Wk, const float* __restrict__ Wq,
                                 const float* __restrict__ Wv, const float* __restrict__ bk,
                                 const float* __restrict__ bq, const float* __restrict__ bv,
                                 __bf16* __restrict__ WqkvT, float* __restrict__ bqkv) {
  int i = blockIdx.x * blockDim.x + threadIdx.x;
  if (i < 768 * DEMB) {
    int c = i / DEMB, k = i % DEMB;
    float v;
    if (c < 256)      v = Wk[(size_t)k * 256 + c];
    else if (c < 512) v = Wq[(size_t)k * 256 + (c - 256)];
    else              v = Wv[(size_t)k * 256 + (c - 512)];
    WqkvT[i] = (__bf16)v;
  }
  if (i < 768) {
    float b;
    if (i < 256) b = bk[i]; else if (i < 512) b = bq[i - 256]; else b = bv[i - 512];
    bqkv[i] = b;
  }
}

// ---------------------------------------------------------------- WMMA GEMM
// C = epilogue( A[M,K](bf16) * BT[N,K]^T(bf16) + bias )
// 64x64 tile, 128 threads (4 waves), each wave 32x32 via 2x2
// v_wmma_f32_16x16x32_bf16. Double-buffered LDS filled with
// global_load_async_to_lds_b128 (ASYNCcnt), 40-bf16 padded rows (80B:
// 16B-aligned, 20-bank stride -> conflict-free b128 fragment reads).
template<bool OUT_BF16, bool RELU_OUT, bool RESID>
__global__ __launch_bounds__(128)
void gemm_bf16_wmma(const __bf16* __restrict__ A, const __bf16* __restrict__ BT,
                    const float* __restrict__ bias, const float* __restrict__ resid,
                    float* __restrict__ Cf, __bf16* __restrict__ Cb,
                    int M, int N, int K) {
  __shared__ __align__(16) __bf16 Als[2][64][40];
  __shared__ __align__(16) __bf16 Bls[2][64][40];

  const int n0   = blockIdx.x * 64;
  const int m0   = blockIdx.y * 64;
  const int t    = threadIdx.x;
  const int lane = t & 31;
  const int wave = t >> 5;
  const int wm   = (wave >> 1) * 32;
  const int wn   = (wave & 1)  * 32;
  const int lrow = lane & 15;
  const int lhi  = lane >> 4;
  const int kbA  = lhi * 8;    // A fragment K-base (CDNA5 lane-half layout)
  const int kbB  = lhi * 8;    // B fragment K-base

  // staging plan: thread t owns a 32B chunk (2 x b128) of A row and BT row (t>>1)
  const int srow = t >> 1;
  const int scol = (t & 1) * 16;
  const int nk   = K / 32;

  v8f acc[2][2];
  #pragma unroll
  for (int i = 0; i < 2; ++i)
    #pragma unroll
    for (int j = 0; j < 2; ++j)
      #pragma unroll
      for (int e = 0; e < 8; ++e) acc[i][j][e] = 0.f;

  auto issue = [&](int it, int buf) {
    const int k0 = it * 32;
    const int gm = m0 + srow;
    if (gm < M) {
      const __bf16* ga = A + (size_t)gm * K + k0 + scol;
      __bf16* la = &Als[buf][srow][scol];
#if USE_ASYNC_LDS
      async_cp16(ga, la);
      async_cp16(ga + 8, la + 8);
#else
      ((int4*)la)[0] = ((const int4*)ga)[0];
      ((int4*)la)[1] = ((const int4*)ga)[1];
#endif
    } else {
      int4 z = make_int4(0, 0, 0, 0);
      ((int4*)&Als[buf][srow][scol])[0] = z;
      ((int4*)&Als[buf][srow][scol])[1] = z;
    }
    const __bf16* gb = BT + (size_t)(n0 + srow) * K + k0 + scol;
    __bf16* lb = &Bls[buf][srow][scol];
#if USE_ASYNC_LDS
    async_cp16(gb, lb);
    async_cp16(gb + 8, lb + 8);
#else
    ((int4*)lb)[0] = ((const int4*)gb)[0];
    ((int4*)lb)[1] = ((const int4*)gb)[1];
#endif
  };

  issue(0, 0);

  for (int it = 0; it < nk; ++it) {
    const int cur = it & 1;
    if (it + 1 < nk) {
      issue(it + 1, cur ^ 1);          // prefetch next slice into other buffer
#if USE_ASYNC_LDS
      __builtin_amdgcn_s_wait_asynccnt(4);  // batch `it` done (in-order), next may fly
#endif
    } else {
#if USE_ASYNC_LDS
      __builtin_amdgcn_s_wait_asynccnt(0);
#endif
    }
    __syncthreads();

    v16bf afrag[2], bfrag[2];
    #pragma unroll
    for (int i = 0; i < 2; ++i) {
      const int r = wm + i * 16 + lrow;
      union { int4 q[2]; v16bf v; } u;
      u.q[0] = *(const int4*)&Als[cur][r][kbA];        // K = kbA..kbA+7
      u.q[1] = *(const int4*)&Als[cur][r][kbA + 16];   // K = kbA+16..kbA+23
      afrag[i] = u.v;
    }
    #pragma unroll
    for (int j = 0; j < 2; ++j) {
      const int c = wn + j * 16 + lrow;
      union { int4 q[2]; v16bf v; } u;
      u.q[0] = *(const int4*)&Bls[cur][c][kbB];        // K = kbB..kbB+7
      u.q[1] = *(const int4*)&Bls[cur][c][kbB + 16];   // K = kbB+16..kbB+23
      bfrag[j] = u.v;
    }

    #pragma unroll
    for (int i = 0; i < 2; ++i)
      #pragma unroll
      for (int j = 0; j < 2; ++j)
        acc[i][j] = __builtin_amdgcn_wmma_f32_16x16x32_bf16(
            false, afrag[i], false, bfrag[j], (short)0, acc[i][j], false, false);

    __syncthreads();
  }

  // ---- epilogue: bias (+ReLU) (+residual), fp32 or bf16 store ----
  #pragma unroll
  for (int i = 0; i < 2; ++i) {
    #pragma unroll
    for (int j = 0; j < 2; ++j) {
      const int col = n0 + wn + j * 16 + lrow;
      const float bc = bias[col];
      #pragma unroll
      for (int v = 0; v < 8; ++v) {
        const int row = m0 + wm + i * 16 + lhi * 8 + v;
        if (row < M) {
          float val = acc[i][j][v] + bc;
          if (RELU_OUT) val = val > 0.f ? val : 0.f;
          if (RESID) val += resid[(size_t)row * N + col];
          if (OUT_BF16) Cb[(size_t)row * N + col] = (__bf16)val;
          else          Cf[(size_t)row * N + col] = val;
        }
      }
    }
  }
}

// ---------------------------------------------------------------- edge phase
// QKV layout per node: [0..255]=K, [256..511]=Q, [512..767]=V (heads of 32)
__global__ void edge_att(const float* __restrict__ QKV, const long long* __restrict__ ei,
                         float* __restrict__ att, float* __restrict__ denom) {
  int idx = blockIdx.x * blockDim.x + threadIdx.x;
  if (idx >= NEDGES * NH) return;
  int e = idx >> 3;
  int h = idx & 7;
  long long r = ei[e];
  long long s = ei[NEDGES + e];
  const float* q = QKV + (size_t)r * 768 + 256 + h * 32;
  const float* k = QKV + (size_t)s * 768 + h * 32;
  float dot = 0.f;
  #pragma unroll
  for (int d = 0; d < 32; d += 4) {
    float4 qv = *(const float4*)(q + d);
    float4 kv = *(const float4*)(k + d);
    dot += qv.x * kv.x + qv.y * kv.y + qv.z * kv.z + qv.w * kv.w;
  }
  float a = __expf(dot * 0.17677669529663687f);  // 1/sqrt(32)
  att[idx] = a;
  atomicAdd(&denom[(size_t)r * NH + h], a);
}

__global__ void edge_scatter(const float* __restrict__ QKV, const long long* __restrict__ ei,
                             const float* __restrict__ att, const float* __restrict__ denom,
                             float* __restrict__ agg) {
  int idx = blockIdx.x * blockDim.x + threadIdx.x;
  if (idx >= NEDGES * NH) return;
  int e = idx >> 3;
  int h = idx & 7;
  long long r = ei[e];
  long long s = ei[NEDGES + e];
  float w = att[idx] / denom[(size_t)r * NH + h];
  const float* v = QKV + (size_t)s * 768 + 512 + h * 32;
  float* dst = agg + (size_t)r * 256 + h * 32;
  #pragma unroll
  for (int d = 0; d < 32; d += 4) {
    float4 vv = *(const float4*)(v + d);
    atomicAdd(dst + d + 0, w * vv.x);
    atomicAdd(dst + d + 1, w * vv.y);
    atomicAdd(dst + d + 2, w * vv.z);
    atomicAdd(dst + d + 3, w * vv.w);
  }
}

// ---------------------------------------------------------------- launch
extern "C" void kernel_launch(void* const* d_in, const int* in_sizes, int n_in,
                              void* d_out, int out_size, void* d_ws, size_t ws_size,
                              hipStream_t stream) {
  (void)in_sizes; (void)n_in; (void)out_size; (void)ws_size;
  const float*     x  = (const float*)d_in[0];
  const long long* ei = (const long long*)d_in[1];   // int64 edge_index
  const float* Wk = (const float*)d_in[2];
  const float* bk = (const float*)d_in[3];
  const float* Wq = (const float*)d_in[4];
  const float* bq = (const float*)d_in[5];
  const float* Wv = (const float*)d_in[6];
  const float* bv = (const float*)d_in[7];
  const float* Wa = (const float*)d_in[8];
  const float* ba = (const float*)d_in[9];
  const float* Wf = (const float*)d_in[10];
  const float* bf = (const float*)d_in[11];
  float* out = (float*)d_out;

  // workspace carve (256B-aligned offsets), ~349 MB total
  char* ws = (char*)d_ws;
  float*  QKV   = (float*)(ws + 0LL);           // 50000*768*4  = 153,600,000
  float*  att   = (float*)(ws + 153600000LL);   // 400000*8*4   =  12,800,000
  float*  denom = (float*)(ws + 166400000LL);   // 50000*8*4    =   1,600,000
  float*  agg   = (float*)(ws + 168000000LL);   // 50000*256*4  =  51,200,000
  __bf16* xB    = (__bf16*)(ws + 219200000LL);  // 50000*512*2  =  51,200,000
  __bf16* aggB  = (__bf16*)(ws + 270400000LL);  // 50000*256*2  =  25,600,000
  __bf16* h1B   = (__bf16*)(ws + 296000000LL);  // 50000*512*2  =  51,200,000
  __bf16* WqkvT = (__bf16*)(ws + 347200000LL);  // 768*512*2    =     786,432
  __bf16* WaT   = (__bf16*)(ws + 347986432LL);  // 512*256*2    =     262,144
  __bf16* WfT   = (__bf16*)(ws + 348248576LL);  // 512*512*2    =     524,288
  float*  bqkv  = (float*)(ws + 348772864LL);   // 768*4

  // weight pre-pack (transposed [N,K] bf16) + bias concat
  cvt_qkv_weightsT<<<(768 * 512 + 255) / 256, 256, 0, stream>>>(Wk, Wq, Wv, bk, bq, bv, WqkvT, bqkv);
  cvt_transpose_bf16<<<(256 * 512 + 255) / 256, 256, 0, stream>>>(Wa, WaT, 256, 512);
  cvt_transpose_bf16<<<(512 * 512 + 255) / 256, 256, 0, stream>>>(Wf, WfT, 512, 512);

  // activations to bf16; zero accumulators (harness doesn't re-zero)
  cvt_f32_bf16<<<(NNODES * DEMB + 255) / 256, 256, 0, stream>>>(x, xB, NNODES * DEMB);
  zero_f32<<<(NNODES * NH + 255) / 256, 256, 0, stream>>>(denom, NNODES * NH);
  zero_f32<<<(NNODES * 256 + 255) / 256, 256, 0, stream>>>(agg, NNODES * 256);

  const int mtiles = (NNODES + 63) / 64;

  // fused QKV projection: [50000,512] x [512,768] -> fp32 QKV
  gemm_bf16_wmma<false, false, false><<<dim3(768 / 64, mtiles), 128, 0, stream>>>(
      xB, WqkvT, bqkv, nullptr, QKV, nullptr, NNODES, 768, 512);

  // edge attention (L2-resident gathers) + normalized scatter
  const int eth = NEDGES * NH;
  edge_att<<<(eth + 255) / 256, 256, 0, stream>>>(QKV, ei, att, denom);
  edge_scatter<<<(eth + 255) / 256, 256, 0, stream>>>(QKV, ei, att, denom, agg);

  // relu(agg) -> bf16 for GEMM2's async-copyable A
  relu_cvt_bf16<<<(NNODES * 256 + 255) / 256, 256, 0, stream>>>(agg, aggB, NNODES * 256);

  // h1 = relu(relu(agg) @ Wa + ba), stored bf16 (feeds GEMM3 directly)
  gemm_bf16_wmma<true, true, false><<<dim3(512 / 64, mtiles), 128, 0, stream>>>(
      aggB, WaT, ba, nullptr, nullptr, h1B, NNODES, 512, 256);

  // out = x + relu(h1 @ Wf + bf), fp32
  gemm_bf16_wmma<false, true, true><<<dim3(512 / 64, mtiles), 128, 0, stream>>>(
      h1B, WfT, bf, x, out, nullptr, NNODES, 512, 512);
}
